// Similarity_34342558499097
// MI455X (gfx1250) — matrix operand order repaired and checked by reading, not compiled, not run on hardware
//
#include <hip/hip_runtime.h>

typedef _Float16 v16h __attribute__((ext_vector_type(16)));
typedef _Float16 v8h  __attribute__((ext_vector_type(8)));
typedef _Float16 v4h  __attribute__((ext_vector_type(4)));
typedef float    v8f  __attribute__((ext_vector_type(8)));
typedef float    v4f  __attribute__((ext_vector_type(4)));

namespace {
constexpr int   kB = 32;    // batch
constexpr int   kN = 16;    // n (mean-reduced)
constexpr int   kT = 256;   // t
constexpr int   kD = 512;   // d (contraction)
constexpr int   kC = 512;   // c
constexpr float kInvTau = 1.0f / 0.07f;   // sig_T_train == sig_T_infer == 0.07
constexpr float kEps    = 1e-12f;

__device__ __forceinline__ float wave_sum32(float v) {
#pragma unroll
    for (int m = 16; m >= 1; m >>= 1) v += __shfl_xor(v, m, 32);
    return v;
}
} // namespace

// ---------------------------------------------------------------------------
// Kernel 1: per-row L2 normalize (if split==1) + mean over n, emit f16 v_mean.
// One wave32 per (b,t) row; lane j-loop covers 512 floats as 4x float4.
// Reads 256MB coalesced (the HBM-bound phase), writes 8MB f16.
// ---------------------------------------------------------------------------
__global__ __launch_bounds__(256) void vmean_norm_f16_kernel(
    const float* __restrict__ v, const int* __restrict__ splitp,
    _Float16* __restrict__ vmean) {
    const int lane = threadIdx.x & 31;
    const int wave = (blockIdx.x * blockDim.x + threadIdx.x) >> 5;
    const int b = wave >> 8;       // kT == 256 waves per batch
    const int t = wave & (kT - 1);
    const bool do_norm = (*splitp == 1);

    float acc[16];
#pragma unroll
    for (int i = 0; i < 16; ++i) acc[i] = 0.0f;

    const float* rowb = v + ((size_t)b * kN * kT + t) * kD;
#pragma unroll 1
    for (int n = 0; n < kN; ++n) {
        const float* p = rowb + (size_t)n * (kT * kD);
        float x[16];
        float ss = 0.0f;
#pragma unroll
        for (int j = 0; j < 4; ++j) {
            v4f q = *(const v4f*)(p + (lane + 32 * j) * 4);  // global_load_b128
#pragma unroll
            for (int e = 0; e < 4; ++e) {
                x[4 * j + e] = q[e];
                ss += q[e] * q[e];
            }
        }
        ss = wave_sum32(ss);
        const float inv = do_norm ? (1.0f / fmaxf(sqrtf(ss), kEps)) : 1.0f;
#pragma unroll
        for (int i = 0; i < 16; ++i) acc[i] += x[i] * inv;
    }

    const float s = 1.0f / (float)kN;
    _Float16* o = vmean + ((size_t)b * kT + t) * kD;
#pragma unroll
    for (int j = 0; j < 4; ++j) {
        v4h h;
#pragma unroll
        for (int e = 0; e < 4; ++e) h[e] = (_Float16)(acc[4 * j + e] * s);
        *(v4h*)(o + (lane + 32 * j) * 4) = h;  // 8B coalesced store
    }
}

// ---------------------------------------------------------------------------
// Kernel 2: normalize t_feat rows (if split==1), emit f16. One wave per row.
// ---------------------------------------------------------------------------
__global__ __launch_bounds__(256) void tnorm_f16_kernel(
    const float* __restrict__ tf, const int* __restrict__ splitp,
    _Float16* __restrict__ out) {
    const int lane = threadIdx.x & 31;
    const int row  = (blockIdx.x * blockDim.x + threadIdx.x) >> 5;  // [0, kC)
    const bool do_norm = (*splitp == 1);

    const float* p = tf + (size_t)row * kD;
    float x[16];
    float ss = 0.0f;
#pragma unroll
    for (int j = 0; j < 4; ++j) {
        v4f q = *(const v4f*)(p + (lane + 32 * j) * 4);
#pragma unroll
        for (int e = 0; e < 4; ++e) {
            x[4 * j + e] = q[e];
            ss += q[e] * q[e];
        }
    }
    ss = wave_sum32(ss);
    const float inv = do_norm ? (1.0f / fmaxf(sqrtf(ss), kEps)) : 1.0f;

    _Float16* o = out + (size_t)row * kD;
#pragma unroll
    for (int j = 0; j < 4; ++j) {
        v4h h;
#pragma unroll
        for (int e = 0; e < 4; ++e) h[e] = (_Float16)(x[4 * j + e] * inv);
        *(v4h*)(o + (lane + 32 * j) * 4) = h;
    }
}

// ---------------------------------------------------------------------------
// Kernel 3: batched GEMM with v_wmma_f32_16x16x32_f16.
//   A = t_norm  (M = c = 512, K = d = 512)   row-major, K contiguous
//   B = v_mean  (N = t = 256, K = d = 512)   row-major, K contiguous
//   D[b, c, t] = (A x B^T) * (1/tau)
// One wave computes a 32(c) x 64(t) tile: 2x4 accumulator fragments,
// 16 K-steps of 8 WMMAs. 64 waves per batch, 2048 waves total.
//
// Fragment per-lane layout (CDNA5 ISA 7.12.2, 16-bit, wave32):
//   A: row = lane&15; halves K: {k+8*hi .. +7} and {k+16+8*hi .. +7}
//   B: col = lane&15; halves K: {k+8*hi .. +7} and {k+16+8*hi .. +7}
//   C: VGPR r, lane l -> M = r + 8*(l>>4), N = l&15
// ---------------------------------------------------------------------------
__global__ __launch_bounds__(256) void sim_wmma_gemm_kernel(
    const _Float16* __restrict__ Tn, const _Float16* __restrict__ Vm,
    float* __restrict__ out) {
    const int lane = threadIdx.x & 31;
    const int wave = blockIdx.x * (blockDim.x >> 5) + (threadIdx.x >> 5);
    const int b    = wave >> 6;          // 64 waves per batch
    const int rem  = wave & 63;
    const int cBase = (rem & 15) * 32;   // 16 tiles over c=512
    const int tBase = (rem >> 4) * 64;   // 4 tiles over t=256

    const int hi   = lane >> 4;          // half-wave select
    const int lrow = lane & 15;

    v8f c[2][4];
#pragma unroll
    for (int i = 0; i < 2; ++i)
#pragma unroll
        for (int j = 0; j < 4; ++j)
#pragma unroll
            for (int r = 0; r < 8; ++r) c[i][j][r] = 0.0f;

    const _Float16* aRow[2];
    aRow[0] = Tn + (size_t)(cBase + lrow) * kD;
    aRow[1] = aRow[0] + 16 * kD;
    const _Float16* bRow[4];
#pragma unroll
    for (int j = 0; j < 4; ++j)
        bRow[j] = Vm + ((size_t)b * kT + tBase + j * 16 + lrow) * kD;

#pragma unroll 2
    for (int k = 0; k < kD; k += 32) {
        union { v16h v; v8h h[2]; } a[2], bb[4];
        const int ka = k + hi * 8;
        const int kb = k + hi * 8;
#pragma unroll
        for (int i = 0; i < 2; ++i) {
            a[i].h[0] = *(const v8h*)(aRow[i] + ka);       // global_load_b128
            a[i].h[1] = *(const v8h*)(aRow[i] + ka + 16);
        }
#pragma unroll
        for (int j = 0; j < 4; ++j) {
            bb[j].h[0] = *(const v8h*)(bRow[j] + kb);
            bb[j].h[1] = *(const v8h*)(bRow[j] + kb + 16);
        }
#pragma unroll
        for (int i = 0; i < 2; ++i)
#pragma unroll
            for (int j = 0; j < 4; ++j)
                c[i][j] = __builtin_amdgcn_wmma_f32_16x16x32_f16(
                    /*neg_a=*/false, a[i].v, /*neg_b=*/false, bb[j].v,
                    /*c_mod=*/(short)0, c[i][j],
                    /*reuse_a=*/false, /*reuse_b=*/false);
    }

    // Store D[b, c, t] * (1/tau); half-wave rows are 64B-contiguous in t.
#pragma unroll
    for (int i = 0; i < 2; ++i) {
#pragma unroll
        for (int j = 0; j < 4; ++j) {
            const int tCol = tBase + j * 16 + lrow;
            float* o = out + ((size_t)b * kC + cBase + i * 16 + 8 * hi) * kT + tCol;
#pragma unroll
            for (int r = 0; r < 8; ++r) o[(size_t)r * kT] = c[i][j][r] * kInvTau;
        }
    }
}

extern "C" void kernel_launch(void* const* d_in, const int* in_sizes, int n_in,
                              void* d_out, int out_size, void* d_ws, size_t ws_size,
                              hipStream_t stream) {
    const float* v_feat = (const float*)d_in[0];  // (32,16,256,512) f32
    const float* t_feat = (const float*)d_in[1];  // (512,512) f32
    const int*   split  = (const int*)d_in[2];    // scalar
    float*       out    = (float*)d_out;          // (32,512,256) f32

    // Workspace: f16 normalized t_feat (512KB) then f16 v_mean (8MB).
    _Float16* t_h = (_Float16*)d_ws;
    _Float16* v_h = t_h + (size_t)kC * kD;

    // Phase 1: streaming reduce/normalize (HBM-bound, ~272MB traffic).
    {
        const int waves = kB * kT;                     // 8192
        vmean_norm_f16_kernel<<<waves / 8, 256, 0, stream>>>(v_feat, split, v_h);
    }
    {
        const int waves = kC;                          // 512
        tnorm_f16_kernel<<<waves / 8, 256, 0, stream>>>(t_feat, split, t_h);
    }

    // Phase 2: 4.3 GFLOP f16-WMMA GEMM, operands L2-resident.
    {
        const int waves = kB * (kC / 32) * (kT / 64);  // 2048
        sim_wmma_gemm_kernel<<<waves / 8, 256, 0, stream>>>(t_h, v_h, out);
    }
}
